// GraphCls_86663850098985
// MI455X (gfx1250) — compile-verified, matrix operand order checked by reading, not run on hardware
//
#include <hip/hip_runtime.h>
#include <hip/hip_bf16.h>

// ---------------- problem constants (match reference) ----------------
#define NNODES  50000      // B * NP
#define NPG     6250       // nodes per graph
#define BGR     8          // graphs
#define CIN     512
#define HD      256        // hidden
#define H3      768        // 3*H
#define NEDGE   800000     // B * EPG
#define KTOP    5000

typedef __attribute__((ext_vector_type(8)))  __bf16 v8bf;
typedef __attribute__((ext_vector_type(16))) __bf16 v16bf;
typedef __attribute__((ext_vector_type(8)))  float  v8f;

__device__ __forceinline__ unsigned short bfbits(float f) {
  union { float f; unsigned u; } c; c.f = f;
  unsigned u = c.u;
  unsigned r = (u + 0x7FFFu + ((u >> 16) & 1u)) >> 16;   // RNE to bf16
  return (unsigned short)r;
}

__device__ __forceinline__ void atomic_add_f32(float* p, float v) {
  __hip_atomic_fetch_add(p, v, __ATOMIC_RELAXED, __HIP_MEMORY_SCOPE_AGENT);
}

// ---------------- WMMA GEMM: C[M,256] = A[M,K] (f32) @ B[K,256] (f32) ------
// A,B converted to bf16 in LDS; accumulate f32 via v_wmma_f32_16x16x32_bf16.
// Block tile 128x128, 8 waves, wave tile 32x64 (2x4 frags), K-step 32.
__global__ __launch_bounds__(256)
void gemm_bf16_kernel(const float* __restrict__ A, const float* __restrict__ Bm,
                      float* __restrict__ C, int M, int K) {
  __shared__ __align__(16) unsigned short Alds[128 * 40]; // 40 ushorts = 80B row stride
  __shared__ __align__(16) unsigned short Blds[128 * 40]; // Bt: [n][k]

  const int tid  = threadIdx.x;
  const int lane = tid & 31;
  const int wv   = tid >> 5;          // 0..7
  const int wr   = wv & 3;            // 4 row groups of 32
  const int wc   = wv >> 2;           // 2 col groups of 64
  const int sel  = lane >> 4;         // 0/1 half-wave (K-halfword select)
  const int l16  = lane & 15;
  const int m0   = blockIdx.x * 128;
  const int n0   = blockIdx.y * 128;

  v8f acc[2][4];
  v8f vzero = {0.f,0.f,0.f,0.f,0.f,0.f,0.f,0.f};
#pragma unroll
  for (int t = 0; t < 2; ++t)
#pragma unroll
    for (int u = 0; u < 4; ++u) acc[t][u] = vzero;

  for (int kk = 0; kk < K; kk += 32) {
    // ---- stage A tile 128x32 (row-major, bf16) ----
#pragma unroll
    for (int i = 0; i < 4; ++i) {
      int idx = tid + i * 256;          // 0..1023
      int row = idx >> 3;
      int kq  = (idx & 7) * 4;
      float4 v = make_float4(0.f, 0.f, 0.f, 0.f);
      int gm = m0 + row;
      if (gm < M) v = *(const float4*)&A[(size_t)gm * K + kk + kq];
      uint2 p;
      p.x = ((unsigned)bfbits(v.y) << 16) | bfbits(v.x);
      p.y = ((unsigned)bfbits(v.w) << 16) | bfbits(v.z);
      *(uint2*)&Alds[row * 40 + kq] = p;
    }
    // ---- stage B tile 32x128 transposed -> Bt[n][k] ----
#pragma unroll
    for (int i = 0; i < 4; ++i) {
      int idx = tid + i * 256;          // 0..1023
      int kr  = idx >> 5;               // 0..31
      int nq  = (idx & 31) * 4;         // 0..124
      float4 v = *(const float4*)&Bm[(size_t)(kk + kr) * HD + n0 + nq];
      Blds[(nq + 0) * 40 + kr] = bfbits(v.x);
      Blds[(nq + 1) * 40 + kr] = bfbits(v.y);
      Blds[(nq + 2) * 40 + kr] = bfbits(v.z);
      Blds[(nq + 3) * 40 + kr] = bfbits(v.w);
    }
    __syncthreads();

    // ---- fragments: lane layout per CDNA5 ISA 16-bit A 16x32 ----
    v16bf afr[2], bfr[4];
#pragma unroll
    for (int t = 0; t < 2; ++t) {
      int row = wr * 32 + t * 16 + l16;
      v8bf lo = *(const v8bf*)&Alds[row * 40 + sel * 8];        // K 0-7 / 8-15
      v8bf hi = *(const v8bf*)&Alds[row * 40 + 16 + sel * 8];   // K 16-23 / 24-31
      afr[t] = __builtin_shufflevector(lo, hi, 0,1,2,3,4,5,6,7,8,9,10,11,12,13,14,15);
    }
#pragma unroll
    for (int u = 0; u < 4; ++u) {
      int col = wc * 64 + u * 16 + l16;
      v8bf lo = *(const v8bf*)&Blds[col * 40 + sel * 8];
      v8bf hi = *(const v8bf*)&Blds[col * 40 + 16 + sel * 8];
      bfr[u] = __builtin_shufflevector(lo, hi, 0,1,2,3,4,5,6,7,8,9,10,11,12,13,14,15);
    }
#pragma unroll
    for (int t = 0; t < 2; ++t)
#pragma unroll
      for (int u = 0; u < 4; ++u)
        acc[t][u] = __builtin_amdgcn_wmma_f32_16x16x32_bf16(
            false, afr[t], false, bfr[u], (short)0, acc[t][u], false, false);
    __syncthreads();
  }

  // ---- epilogue: D layout — VGPR j -> M=j / j+8 (lane half), N = lane%16 ----
#pragma unroll
  for (int t = 0; t < 2; ++t) {
#pragma unroll
    for (int u = 0; u < 4; ++u) {
      int gn = n0 + wc * 64 + u * 16 + l16;
#pragma unroll
      for (int j = 0; j < 8; ++j) {
        int gm = m0 + wr * 32 + t * 16 + j + sel * 8;
        if (gm < M) C[(size_t)gm * HD + gn] = acc[t][u][j];
      }
    }
  }
}

// ---------------- small kernels ----------------
__global__ void zero_kernel(float* p, int n) {
  int i = blockIdx.x * blockDim.x + threadIdx.x;
  if (i < n) p[i] = 0.f;
}

__global__ void deg_accum_kernel(const int* __restrict__ dst,
                                 const float* __restrict__ ew, float* deg, int E) {
  int e = blockIdx.x * blockDim.x + threadIdx.x;
  if (e < E) atomic_add_f32(&deg[dst[e]], ew[e]);
}

__global__ void dinv_kernel(float* deg_inplace, int n) {
  int i = blockIdx.x * blockDim.x + threadIdx.x;
  if (i < n) deg_inplace[i] = rsqrtf(deg_inplace[i] + 1.0f);
}

__global__ void norm_kernel(const int* __restrict__ src, const int* __restrict__ dst,
                            const float* __restrict__ ew, const float* __restrict__ dinv,
                            float* __restrict__ nrm, int E) {
  int e = blockIdx.x * blockDim.x + threadIdx.x;
  if (e < E) nrm[e] = dinv[src[e]] * ew[e] * dinv[dst[e]];
}

// out[n,c] = dinv[n]^2 * xw[n,c] + b[c]   (self-loop term + bias, pre-scatter)
__global__ __launch_bounds__(256)
void init_out_kernel(const float* __restrict__ xw, const float* __restrict__ dinv,
                     const float* __restrict__ bias, float* __restrict__ out) {
  int n = blockIdx.x, c = threadIdx.x;
  float di = dinv[n];
  out[(size_t)n * HD + c] = di * di * xw[(size_t)n * HD + c] + bias[c];
}

// wave-per-edge scatter: out[dst] += nrm[e] * xw[src]  (8 ch / lane, float4 x2)
__global__ __launch_bounds__(256)
void scatter_kernel(const int* __restrict__ src, const int* __restrict__ dst,
                    const float* __restrict__ nrm, const float* __restrict__ xw,
                    float* __restrict__ out, int E) {
  int e = blockIdx.x * 8 + (threadIdx.x >> 5);
  if (e >= E) return;
  int lane = threadIdx.x & 31;
  int s = src[e], d = dst[e];
  float w = nrm[e];
  const float4* ps = (const float4*)&xw[(size_t)s * HD + lane * 8];
  float4 a = ps[0], b = ps[1];
  float* po = &out[(size_t)d * HD + lane * 8];
  atomic_add_f32(po + 0, w * a.x); atomic_add_f32(po + 1, w * a.y);
  atomic_add_f32(po + 2, w * a.z); atomic_add_f32(po + 3, w * a.w);
  atomic_add_f32(po + 4, w * b.x); atomic_add_f32(po + 5, w * b.y);
  atomic_add_f32(po + 6, w * b.z); atomic_add_f32(po + 7, w * b.w);
}

__global__ __launch_bounds__(256)
void relu_kernel(float* __restrict__ x) {
  size_t i = (size_t)blockIdx.x * HD + threadIdx.x;
  x[i] = fmaxf(x[i], 0.f);
}

__global__ void pwnorm_kernel(const float* __restrict__ pw, float* pwn) {
  __shared__ float red[256];
  int tid = threadIdx.x;
  float a = 0.f;
  for (int i = tid; i < H3; i += 256) a += pw[i] * pw[i];
  red[tid] = a; __syncthreads();
  for (int s = 128; s > 0; s >>= 1) { if (tid < s) red[tid] += red[tid + s]; __syncthreads(); }
  if (tid == 0) pwn[0] = sqrtf(red[0]);
}

// wave-per-node score = sigmoid(dot(cat_x, pw)/||pw||)
__global__ __launch_bounds__(256)
void score_kernel(const float* __restrict__ x1, const float* __restrict__ x2,
                  const float* __restrict__ x3, const float* __restrict__ pw,
                  const float* __restrict__ pwn, float* __restrict__ score) {
  int n = blockIdx.x * 8 + (threadIdx.x >> 5);
  int lane = threadIdx.x & 31;
  size_t base = (size_t)n * HD;
  float acc = 0.f;
  for (int i = lane; i < HD; i += 32)
    acc += x1[base + i] * pw[i] + x2[base + i] * pw[HD + i] + x3[base + i] * pw[2 * HD + i];
  for (int o = 16; o > 0; o >>= 1) acc += __shfl_xor(acc, o, 32);
  if (lane == 0) {
    float z = acc / pwn[0];
    score[n] = 1.f / (1.f + __expf(-z));
  }
}

// per-graph K-th-largest threshold via bit-level binary search (scores in (0,1))
__global__ void topk_thresh_kernel(const float* __restrict__ score,
                                   unsigned* theta, int* need_eq, int* ticket) {
  int g = blockIdx.x, tid = threadIdx.x;
  const unsigned* sb = (const unsigned*)(score + (size_t)g * NPG);
  __shared__ int cnt;
  unsigned lo = 0u, hi = 0x3F800001u;
  while (hi - lo > 1u) {
    unsigned mid = lo + ((hi - lo) >> 1);
    if (tid == 0) cnt = 0;
    __syncthreads();
    int c = 0;
    for (int i = tid; i < NPG; i += 256) c += (sb[i] >= mid) ? 1 : 0;
    atomicAdd(&cnt, c);
    __syncthreads();
    int tot = cnt;
    __syncthreads();
    if (tot >= KTOP) lo = mid; else hi = mid;
  }
  if (tid == 0) cnt = 0;
  __syncthreads();
  int c = 0;
  for (int i = tid; i < NPG; i += 256) c += (sb[i] > lo) ? 1 : 0;
  atomicAdd(&cnt, c);
  __syncthreads();
  if (tid == 0) { theta[g] = lo; need_eq[g] = KTOP - cnt; ticket[g] = 0; }
}

__global__ void select_kernel(const float* __restrict__ score, const unsigned* __restrict__ theta,
                              const int* __restrict__ need_eq, int* ticket,
                              float* __restrict__ ssel, int n_total) {
  int n = blockIdx.x * blockDim.x + threadIdx.x;
  if (n >= n_total) return;
  int g = n / NPG;
  union { float f; unsigned u; } c; c.f = score[n];
  float out = 0.f;
  if (c.u > theta[g]) out = c.f;
  else if (c.u == theta[g]) { if (atomicAdd(&ticket[g], 1) < need_eq[g]) out = c.f; }
  ssel[n] = out;
}

// per (graph, 256-channel chunk): mean & max over selected of score*x
__global__ __launch_bounds__(256)
void readout_kernel(const float* __restrict__ x1, const float* __restrict__ x2,
                    const float* __restrict__ x3, const float* __restrict__ ssel,
                    float* __restrict__ rout) {
  int g = blockIdx.x / 3, ch = blockIdx.x % 3;
  const float* seg = (ch == 0) ? x1 : ((ch == 1) ? x2 : x3);
  int tid = threadIdx.x;
  __shared__ float sl[256];
  float sum = 0.f, mx = -1e30f;
  for (int base = 0; base < NPG; base += 256) {
    __syncthreads();
    int nn = base + tid;
    sl[tid] = (nn < NPG) ? ssel[(size_t)g * NPG + nn] : 0.f;
    __syncthreads();
    int lim = (NPG - base < 256) ? (NPG - base) : 256;
    for (int j = 0; j < lim; ++j) {
      float s = sl[j];
      if (s > 0.f) {
        float v = s * seg[((size_t)g * NPG + base + j) * HD + tid];
        sum += v;
        mx = fmaxf(mx, v);
      }
    }
  }
  rout[(size_t)g * (2 * H3) + ch * 256 + tid] = sum * (1.f / (float)KTOP);
  rout[(size_t)g * (2 * H3) + H3 + ch * 256 + tid] = mx;
}

__global__ __launch_bounds__(256)
void mlp_kernel(const float* __restrict__ rout,
                const float* __restrict__ lw1, const float* __restrict__ lb1,
                const float* __restrict__ lw2, const float* __restrict__ lb2,
                const float* __restrict__ lw3, const float* __restrict__ lb3,
                float* __restrict__ out) {
  int g = blockIdx.x, tid = threadIdx.x;
  __shared__ float r[2 * H3];
  __shared__ float h1[HD];
  __shared__ float h2[HD / 2];
  for (int i = tid; i < 2 * H3; i += 256) r[i] = rout[(size_t)g * (2 * H3) + i];
  __syncthreads();
  float a = lb1[tid];
  for (int k = 0; k < 2 * H3; ++k) a += r[k] * lw1[(size_t)k * HD + tid];
  h1[tid] = fmaxf(a, 0.f);
  __syncthreads();
  if (tid < HD / 2) {
    float b = lb2[tid];
    for (int k = 0; k < HD; ++k) b += h1[k] * lw2[(size_t)k * (HD / 2) + tid];
    h2[tid] = fmaxf(b, 0.f);
  }
  __syncthreads();
  if (tid < 3) {
    float c = lb3[tid];
    for (int k = 0; k < HD / 2; ++k) c += h2[k] * lw3[(size_t)k * 3 + tid];
    out[g * 3 + tid] = c;
  }
}

// ---------------- launch ----------------
extern "C" void kernel_launch(void* const* d_in, const int* in_sizes, int n_in,
                              void* d_out, int out_size, void* d_ws, size_t ws_size,
                              hipStream_t stream) {
  (void)in_sizes; (void)n_in; (void)out_size; (void)ws_size;
  const float* x   = (const float*)d_in[0];
  const int*   ei  = (const int*)d_in[1];     // [2, E]
  const float* ea  = (const float*)d_in[2];   // [E]
  const float* W1  = (const float*)d_in[4];
  const float* b1  = (const float*)d_in[5];
  const float* W2  = (const float*)d_in[6];
  const float* b2  = (const float*)d_in[7];
  const float* pw  = (const float*)d_in[8];
  const float* lw1 = (const float*)d_in[9];
  const float* lb1 = (const float*)d_in[10];
  const float* lw2 = (const float*)d_in[11];
  const float* lb2 = (const float*)d_in[12];
  const float* lw3 = (const float*)d_in[13];
  const float* lb3 = (const float*)d_in[14];
  const int* src = ei;
  const int* dst = ei + NEDGE;

  // workspace layout
  size_t NH = (size_t)NNODES * HD;            // 12.8M floats
  float* xw    = (float*)d_ws;
  float* x1    = xw + NH;
  float* x2    = x1 + NH;
  float* x3    = x2 + NH;
  float* dinv  = x3 + NH;                     // NNODES (deg -> dinv in place)
  float* nrm   = dinv + NNODES;               // NEDGE
  float* score = nrm + NEDGE;                 // NNODES
  float* ssel  = score + NNODES;              // NNODES
  float* rout  = ssel + NNODES;               // 8*1536
  float* pwn   = rout + (size_t)BGR * 2 * H3; // 1 (+pad)
  unsigned* theta = (unsigned*)(pwn + 4);
  int* neede = (int*)(theta + 8);
  int* ticket = neede + 8;

  dim3 blk(256);
  // degree -> dinv, edge norms
  zero_kernel<<<(NNODES + 255) / 256, blk, 0, stream>>>(dinv, NNODES);
  deg_accum_kernel<<<(NEDGE + 255) / 256, blk, 0, stream>>>(dst, ea, dinv, NEDGE);
  dinv_kernel<<<(NNODES + 255) / 256, blk, 0, stream>>>(dinv, NNODES);
  norm_kernel<<<(NEDGE + 255) / 256, blk, 0, stream>>>(src, dst, ea, dinv, nrm, NEDGE);
  pwnorm_kernel<<<1, blk, 0, stream>>>(pw, pwn);

  dim3 ggrid((NNODES + 127) / 128, HD / 128);
  int sblocks = (NEDGE + 7) / 8;

  // layer 1: x @ W1 (K=512)
  gemm_bf16_kernel<<<ggrid, blk, 0, stream>>>(x, W1, xw, NNODES, CIN);
  init_out_kernel<<<NNODES, blk, 0, stream>>>(xw, dinv, b1, x1);
  scatter_kernel<<<sblocks, blk, 0, stream>>>(src, dst, nrm, xw, x1, NEDGE);
  relu_kernel<<<NNODES, blk, 0, stream>>>(x1);

  // layer 2: x1 @ W2 (K=256)
  gemm_bf16_kernel<<<ggrid, blk, 0, stream>>>(x1, W2, xw, NNODES, HD);
  init_out_kernel<<<NNODES, blk, 0, stream>>>(xw, dinv, b2, x2);
  scatter_kernel<<<sblocks, blk, 0, stream>>>(src, dst, nrm, xw, x2, NEDGE);
  relu_kernel<<<NNODES, blk, 0, stream>>>(x2);

  // layer 3: x2 @ W2 (K=256)
  gemm_bf16_kernel<<<ggrid, blk, 0, stream>>>(x2, W2, xw, NNODES, HD);
  init_out_kernel<<<NNODES, blk, 0, stream>>>(xw, dinv, b2, x3);
  scatter_kernel<<<sblocks, blk, 0, stream>>>(src, dst, nrm, xw, x3, NEDGE);
  relu_kernel<<<NNODES, blk, 0, stream>>>(x3);

  // pooling + readout + head
  score_kernel<<<NNODES / 8, blk, 0, stream>>>(x1, x2, x3, pw, pwn, score);
  topk_thresh_kernel<<<BGR, blk, 0, stream>>>(score, theta, neede, ticket);
  select_kernel<<<(NNODES + 255) / 256, blk, 0, stream>>>(score, theta, neede, ticket, ssel, NNODES);
  readout_kernel<<<BGR * 3, blk, 0, stream>>>(x1, x2, x3, ssel, rout);
  mlp_kernel<<<BGR, blk, 0, stream>>>(rout, lw1, lb1, lw2, lb2, lw3, lb3, (float*)d_out);
}
